// LSTM_16312285791045
// MI455X (gfx1250) — matrix operand order repaired and checked by reading, not compiled, not run on hardware
//
#include <hip/hip_runtime.h>
#include <hip/hip_bf16.h>

// LSTM on MI455X (gfx1250): bf16 WMMA GEMMs + persistent recurrent kernel.
// B=32, T=2048, D=512, H=512.

typedef __bf16 bf16;
typedef __attribute__((ext_vector_type(16))) __bf16 v16bf;
typedef __attribute__((ext_vector_type(8)))  __bf16 bf16x8;
typedef __attribute__((ext_vector_type(8)))  float  v8f;
typedef __attribute__((ext_vector_type(4)))  float  f32x4;

#define B_  32
#define T_  2048
#define D_  512
#define H_  512
#define NBLK_REC 64   // 2 m-tiles * 32 j-chunks

__device__ __forceinline__ v16bf ld_frag16(const bf16* p0, const bf16* p1) {
    union { v16bf v; bf16x8 h[2]; } u;
    u.h[0] = *reinterpret_cast<const bf16x8*>(p0);
    u.h[1] = *reinterpret_cast<const bf16x8*>(p1);
    return u.v;
}

// ---------------------------------------------------------------------------
// Prep: transpose + convert weights to bf16.
//   WpT  [H=512 out][D=512 in]   : WpT[n*512+k]        = W_p[k*512+n]
//   WallT[2048 out][1024 in]     : WallT[(g*512+j)*1024+k] = W_g[k*512+j]
//   gate order g: 0=f, 1=i, 2=c, 3=o
// ---------------------------------------------------------------------------
__global__ void lstm_prep_kernel(const float* __restrict__ Wp,
                                 const float* __restrict__ Wf,
                                 const float* __restrict__ Wi,
                                 const float* __restrict__ Wc,
                                 const float* __restrict__ Wo,
                                 bf16* __restrict__ WpT,
                                 bf16* __restrict__ WallT) {
    int idx = blockIdx.x * blockDim.x + threadIdx.x;
    if (idx < D_ * H_) {
        int n = idx / D_, k = idx % D_;
        WpT[idx] = (bf16)Wp[(size_t)k * H_ + n];
    }
    if (idx < 2048 * 1024) {
        int n = idx >> 10;          // output column (gate-major)
        int k = idx & 1023;         // input feature
        int g = n >> 9;
        int j = n & 511;
        const float* Wg = (g == 0) ? Wf : (g == 1) ? Wi : (g == 2) ? Wc : Wo;
        WallT[idx] = (bf16)Wg[(size_t)k * H_ + j];
    }
}

// Zero h state (both buffers) and grid-barrier words every launch.
__global__ void lstm_init_kernel(bf16* __restrict__ hbf, int* __restrict__ bar) {
    int idx = blockIdx.x * blockDim.x + threadIdx.x;
    if (idx < 2 * B_ * H_) hbf[idx] = (bf16)0.0f;
    if (idx < 8) bar[idx] = 0;
}

// ---------------------------------------------------------------------------
// Phase 1: proj = tanh(X @ W_p + b_p)  -> bf16 [B*T, H]
// One 16x16 output tile per wave iteration; K = 512 in 16 WMMA steps.
// ---------------------------------------------------------------------------
__global__ void lstm_proj_kernel(const float* __restrict__ X,
                                 const bf16*  __restrict__ WpT,
                                 const float* __restrict__ bp,
                                 bf16* __restrict__ proj) {
    const int lane  = threadIdx.x & 31;
    const int wave  = threadIdx.x >> 5;
    const int wpb   = blockDim.x >> 5;
    const int gwave = blockIdx.x * wpb + wave;
    const int nwave = gridDim.x * wpb;
    const int half  = lane >> 4;
    const int l16   = lane & 15;
    const int MT = (B_ * T_) / 16;   // 4096
    const int NT = H_ / 16;          // 32

    for (int tile = gwave; tile < MT * NT; tile += nwave) {
        const int mt   = tile / NT;
        const int nt   = tile % NT;
        const int row  = mt * 16 + l16;   // A row (global m)
        const int ncol = nt * 16 + l16;   // B column / output column
        const float* xrow = X   + (size_t)row  * D_;
        const bf16*  wcol = WpT + (size_t)ncol * D_;

        v8f acc = {};
        for (int kk = 0; kk < D_ / 32; ++kk) {
            const int ka = kk * 32 + half * 8;    // A: runs [ka..ka+7], [ka+16..ka+23]
            const int kb = kk * 32 + half * 16;   // unused; B is loaded with A's runs at ka
            union { v16bf v; bf16 e[16]; } a;
            f32x4 x0 = *reinterpret_cast<const f32x4*>(xrow + ka);
            f32x4 x1 = *reinterpret_cast<const f32x4*>(xrow + ka + 4);
            f32x4 x2 = *reinterpret_cast<const f32x4*>(xrow + ka + 16);
            f32x4 x3 = *reinterpret_cast<const f32x4*>(xrow + ka + 20);
#pragma unroll
            for (int e = 0; e < 4; ++e) {
                a.e[e]      = (bf16)x0[e];
                a.e[4 + e]  = (bf16)x1[e];
                a.e[8 + e]  = (bf16)x2[e];
                a.e[12 + e] = (bf16)x3[e];
            }
            v16bf bfr = ld_frag16(wcol + ka, wcol + ka + 16);
            acc = __builtin_amdgcn_wmma_f32_16x16x32_bf16(
                      false, a.v, false, bfr, (short)0, acc, false, false);
        }

        const float bias = bp[ncol];
#pragma unroll
        for (int e = 0; e < 8; ++e) {
            float v = tanhf(acc[e] + bias);
            int r = mt * 16 + half * 8 + e;      // D row (global m)
            proj[(size_t)r * H_ + ncol] = (bf16)v;
        }
    }
}

// ---------------------------------------------------------------------------
// Phase 2: persistent recurrent kernel. 64 blocks x 128 threads (4 waves).
// Block (mt, jc): wave g computes gate g's 16x16 tile at rows mt*16..+15,
// cols jc*16..+15.  K = 1024 = [p_t (512) | h (512)] in 32 WMMA steps.
// c kept in registers; h double-buffered bf16 in ws; 1 grid barrier / step.
// ---------------------------------------------------------------------------
__global__ void lstm_rec_kernel(const bf16* __restrict__ proj,
                                const bf16* __restrict__ WallT,
                                const float* __restrict__ bfv,
                                const float* __restrict__ biv,
                                const float* __restrict__ bcv,
                                const float* __restrict__ bov,
                                bf16*  __restrict__ hbf,     // 2 * [B,H]
                                float* __restrict__ out,     // [B,T,H]
                                int*   __restrict__ bar) {
    __shared__ float lds[4][16][16];

    const int tid  = threadIdx.x;
    const int lane = tid & 31;
    const int g    = tid >> 5;           // gate: 0=f 1=i 2=cg 3=o
    const int l16  = lane & 15;
    const int half = lane >> 4;
    const int mt   = blockIdx.x & 1;
    const int jc   = blockIdx.x >> 1;    // 0..31
    const int nblk = gridDim.x;

    const int   arow  = mt * 16 + l16;                 // batch row for A
    const int   ncol  = g * 512 + jc * 16 + l16;       // W_all_T row
    const bf16* wbase = WallT + (size_t)ncol * 1024;
    const float* bp   = (g == 0) ? bfv : (g == 1) ? biv : (g == 2) ? bcv : bov;
    const float bias  = bp[jc * 16 + l16];

    // Elementwise mapping: 2 cells per thread, fixed (b, j) for all t.
    const int e0 = tid * 2, e1 = e0 + 1;
    const int er0 = e0 >> 4, ec0 = e0 & 15;
    const int er1 = e1 >> 4, ec1 = e1 & 15;
    const int b0 = mt * 16 + er0, j0 = jc * 16 + ec0;
    const int b1 = mt * 16 + er1, j1 = jc * 16 + ec1;
    float c0 = 0.0f, c1 = 0.0f;

    for (int t = 0; t < T_; ++t) {
        const bf16* hread  = hbf + (size_t)(t & 1) * (B_ * H_);
        bf16*       hwrite = hbf + (size_t)((t + 1) & 1) * (B_ * H_);
        const bf16* prow   = proj + ((size_t)arow * T_ + t) * H_;
        const bf16* hrow   = hread + (size_t)arow * H_;

        v8f acc = {};
#pragma unroll 4
        for (int kk = 0; kk < 32; ++kk) {
            const int klo = kk * 32 + half * 8;
            v16bf afrag = (kk < 16)
                ? ld_frag16(prow + klo,       prow + klo + 16)
                : ld_frag16(hrow + klo - 512, hrow + klo - 512 + 16);
            const int kb = kk * 32 + half * 16;
            v16bf bfrag = ld_frag16(wbase + klo, wbase + klo + 16);
            acc = __builtin_amdgcn_wmma_f32_16x16x32_bf16(
                      false, afrag, false, bfrag, (short)0, acc, false, false);
        }

        // Activations -> LDS (D layout: row = half*8+e, col = l16)
#pragma unroll
        for (int e = 0; e < 8; ++e) {
            float v = acc[e] + bias;
            v = (g == 2) ? tanhf(v) : (1.0f / (1.0f + __expf(-v)));
            lds[g][half * 8 + e][l16] = v;
        }
        __syncthreads();

        // Cell update: c = f*c + i*cg ; h = o*tanh(c)
        {
            float f0 = lds[0][er0][ec0], i0 = lds[1][er0][ec0];
            float g0 = lds[2][er0][ec0], o0 = lds[3][er0][ec0];
            float f1 = lds[0][er1][ec1], i1 = lds[1][er1][ec1];
            float g1 = lds[2][er1][ec1], o1 = lds[3][er1][ec1];
            c0 = f0 * c0 + i0 * g0;
            c1 = f1 * c1 + i1 * g1;
            float h0 = o0 * tanhf(c0);
            float h1 = o1 * tanhf(c1);
            out[((size_t)b0 * T_ + t) * H_ + j0] = h0;
            out[((size_t)b1 * T_ + t) * H_ + j1] = h1;
            hwrite[b0 * H_ + j0] = (bf16)h0;
            hwrite[b1 * H_ + j1] = (bf16)h1;
        }
        if (t + 1 < T_) __builtin_prefetch(prow + H_, 0, 1);  // next step's p_t row
        __syncthreads();   // LDS reuse guard

        // Grid-wide sense barrier (makes hwrite visible device-wide)
        if (tid == 0) {
            __threadfence();
            int a = __hip_atomic_fetch_add(&bar[0], 1, __ATOMIC_ACQ_REL,
                                           __HIP_MEMORY_SCOPE_AGENT);
            if (a == nblk - 1) {
                bar[0] = 0;
                __hip_atomic_store(&bar[1], t + 1, __ATOMIC_RELEASE,
                                   __HIP_MEMORY_SCOPE_AGENT);
            } else {
                while (__hip_atomic_load(&bar[1], __ATOMIC_ACQUIRE,
                                         __HIP_MEMORY_SCOPE_AGENT) <= t) {
                    __builtin_amdgcn_s_sleep(8);
                }
            }
        }
        __syncthreads();
    }
}

// ---------------------------------------------------------------------------
extern "C" void kernel_launch(void* const* d_in, const int* in_sizes, int n_in,
                              void* d_out, int out_size, void* d_ws, size_t ws_size,
                              hipStream_t stream) {
    const float* X   = (const float*)d_in[0];
    const float* Wp  = (const float*)d_in[1];
    const float* bp  = (const float*)d_in[2];
    const float* Wf  = (const float*)d_in[3];
    const float* bfv = (const float*)d_in[4];
    const float* Wi  = (const float*)d_in[5];
    const float* biv = (const float*)d_in[6];
    const float* Wc  = (const float*)d_in[7];
    const float* bcv = (const float*)d_in[8];
    const float* Wo  = (const float*)d_in[9];
    const float* bov = (const float*)d_in[10];
    float* out = (float*)d_out;

    // Workspace layout (all offsets 256B aligned).
    char* ws = (char*)d_ws;
    size_t off = 0;
    bf16* proj  = (bf16*)(ws + off); off += (size_t)B_ * T_ * H_ * sizeof(bf16); // 64 MB
    bf16* WallT = (bf16*)(ws + off); off += (size_t)2048 * 1024 * sizeof(bf16);  //  4 MB
    bf16* WpT   = (bf16*)(ws + off); off += (size_t)D_ * H_ * sizeof(bf16);      // .5 MB
    bf16* hbf   = (bf16*)(ws + off); off += (size_t)2 * B_ * H_ * sizeof(bf16);  // 64 KB
    int*  bar   = (int*) (ws + off); off += 256;
    (void)ws_size; (void)in_sizes; (void)n_in; (void)out_size;

    // 1) weight transpose/convert (2048*1024 elements dominant)
    lstm_prep_kernel<<<(2048 * 1024 + 255) / 256, 256, 0, stream>>>(
        Wp, Wf, Wi, Wc, Wo, WpT, WallT);
    // 2) zero h state + barrier
    lstm_init_kernel<<<(2 * B_ * H_ + 255) / 256, 256, 0, stream>>>(hbf, bar);
    // 3) parallel projection GEMM + tanh
    lstm_proj_kernel<<<1024, 256, 0, stream>>>(X, WpT, bp, proj);
    // 4) persistent recurrent scan
    lstm_rec_kernel<<<NBLK_REC, 128, 0, stream>>>(
        proj, WallT, bfv, biv, bcv, bov, hbf, out, bar);
}
